// SSloss_76089640616005
// MI455X (gfx1250) — compile-verified, matrix operand order checked by reading, not run on hardware
//
#include <hip/hip_runtime.h>

typedef __attribute__((ext_vector_type(16))) _Float16 v16h;
typedef __attribute__((ext_vector_type(8)))  float    v8f;

#define KGRP   200   // group size (softmax width)
#define DDIM   512   // feature dim
#define NTILES 13    // ceil(200/16) M-tiles of key rows

// One block (128 threads = 4 wave32) per sample n.
// logits[n,k] = dot(query[n,:], key[n*K + k, :]) via v_wmma_f32_16x16x32_f16,
// then log-softmax + weighted mean -> per_sample[n].
__global__ __launch_bounds__(128) void ssloss_persample(
    const float* __restrict__ query,
    const float* __restrict__ key,
    const float* __restrict__ target,
    float* __restrict__ per_sample)
{
    __shared__ float s_query[DDIM];          // 2 KB: this sample's query row
    __shared__ float s_logits[NTILES * 16];  // 208 slots, 200 valid (200..207 = junk, never read)
    __shared__ float sA[128], sB[128], sC[128];

    const int n    = blockIdx.x;
    const int tid  = threadIdx.x;
    const int lane = tid & 31;
    const int wv   = tid >> 5;      // wave id 0..3
    const int m    = lane & 15;     // A-matrix row within tile
    const int h    = lane >> 4;     // k-striping half (ISA 16-bit A layout)

    // Stage query row into LDS once.
    const float* qrow = query + (size_t)n * DDIM;
    for (int i = tid; i < DDIM; i += 128) s_query[i] = qrow[i];
    __syncthreads();

    // Wave w owns tiles w, w+4, w+8 (and w+12 for wave 0). Wave-uniform count.
    const int tcount = (wv == 0) ? 4 : 3;

    const float* rows[4];
    #pragma unroll
    for (int i = 0; i < 4; ++i) {
        int tile = wv + 4 * i;
        int krow = tile * 16 + m;
        if (krow > KGRP - 1) krow = KGRP - 1;   // clamp padded rows (discarded below)
        rows[i] = key + ((size_t)n * KGRP + krow) * DDIM;
    }

    v8f acc[4] = {};

    for (int d0 = 0; d0 < DDIM; d0 += 32) {
        // B tile (32x16 f16): query chunk replicated across all 16 columns,
        // built ONCE per d0 and reused by all tiles this wave owns.
        const float* q = s_query + d0 + h * 8;
        v16h b;
        #pragma unroll
        for (int j = 0; j < 8; ++j) {
            b[j]     = (_Float16)q[j];
            b[8 + j] = (_Float16)q[16 + j];
        }

        const bool pf = (d0 + 32 < DDIM);

        #pragma unroll
        for (int i = 0; i < 4; ++i) {
            if (i < tcount) {                   // wave-uniform scalar branch
                const float* arow = rows[i];
                if (pf)
                    __builtin_prefetch(arow + d0 + 32 + h * 8, 0, 3); // global_prefetch_b8

                // A tile (16x32 f16): lane holds row m; K runs [8h..8h+7], [16+8h..16+8h+7]
                const float4 f0 = *(const float4*)(arow + d0 + h * 8);
                const float4 f1 = *(const float4*)(arow + d0 + h * 8 + 4);
                const float4 f2 = *(const float4*)(arow + d0 + 16 + h * 8);
                const float4 f3 = *(const float4*)(arow + d0 + 16 + h * 8 + 4);
                v16h a;
                a[0]  = (_Float16)f0.x; a[1]  = (_Float16)f0.y;
                a[2]  = (_Float16)f0.z; a[3]  = (_Float16)f0.w;
                a[4]  = (_Float16)f1.x; a[5]  = (_Float16)f1.y;
                a[6]  = (_Float16)f1.z; a[7]  = (_Float16)f1.w;
                a[8]  = (_Float16)f2.x; a[9]  = (_Float16)f2.y;
                a[10] = (_Float16)f2.z; a[11] = (_Float16)f2.w;
                a[12] = (_Float16)f3.x; a[13] = (_Float16)f3.y;
                a[14] = (_Float16)f3.z; a[15] = (_Float16)f3.w;

                acc[i] = __builtin_amdgcn_wmma_f32_16x16x32_f16(
                    /*neg_a=*/false, a, /*neg_b=*/false, b,
                    /*c_mod=*/(short)0, acc[i], /*reuse_a=*/false, /*reuse_b=*/false);
            }
        }
    }

    // D layout: lanes 0-15 hold M=0..7 in v0..7 (N=lane), lanes 16-31 hold M=8..15.
    // Column-0 lanes (m==0 -> lanes 0 and 16) write logits. Max index 207 < 208:
    // unconditional stores, no per-element EXEC juggling.
    #pragma unroll
    for (int i = 0; i < 4; ++i) {
        if (i < tcount && m == 0) {
            const int base = (wv + 4 * i) * 16 + h * 8;
            #pragma unroll
            for (int v = 0; v < 8; ++v) s_logits[base + v] = acc[i][v];
        }
    }
    __syncthreads();

    // --- deterministic log-softmax + weighted loss over K=200 ---
    float lmax = -3.402823466e38f;
    for (int k = tid; k < KGRP; k += 128) lmax = fmaxf(lmax, s_logits[k]);
    sA[tid] = lmax; __syncthreads();
    for (int o = 64; o > 0; o >>= 1) {
        if (tid < o) sA[tid] = fmaxf(sA[tid], sA[tid + o]);
        __syncthreads();
    }
    const float mx = sA[0];
    __syncthreads();

    const float* trow = target + (size_t)n * KGRP;
    float s1 = 0.f, s2 = 0.f, s3 = 0.f;
    for (int k = tid; k < KGRP; k += 128) {
        float l = s_logits[k] - mx;
        float t = trow[k];
        s1 += expf(l);      // sum exp
        s2 += t * l;        // sum t*(logit-max)
        s3 += t;            // sum t
    }
    sA[tid] = s1; sB[tid] = s2; sC[tid] = s3; __syncthreads();
    for (int o = 64; o > 0; o >>= 1) {
        if (tid < o) {
            sA[tid] += sA[tid + o];
            sB[tid] += sB[tid + o];
            sC[tid] += sC[tid + o];
        }
        __syncthreads();
    }
    if (tid == 0) {
        // -(1/K) * sum_k t_k*(l_k - mx - log(sum exp))
        per_sample[n] = -(sB[0] - sC[0] * logf(sA[0])) / (float)KGRP;
    }
}

// Deterministic single-block mean over n per-sample losses.
__global__ __launch_bounds__(256) void ssloss_reduce_mean(
    const float* __restrict__ ws, float* __restrict__ out, int n)
{
    __shared__ float s[256];
    float acc = 0.f;
    for (int i = threadIdx.x; i < n; i += 256) acc += ws[i];
    s[threadIdx.x] = acc; __syncthreads();
    for (int o = 128; o > 0; o >>= 1) {
        if (threadIdx.x < o) s[threadIdx.x] += s[threadIdx.x + o];
        __syncthreads();
    }
    if (threadIdx.x == 0) out[0] = s[0] / (float)n;
}

extern "C" void kernel_launch(void* const* d_in, const int* in_sizes, int n_in,
                              void* d_out, int out_size, void* d_ws, size_t ws_size,
                              hipStream_t stream) {
    const float* query  = (const float*)d_in[0];   // [n, 512]
    const float* key    = (const float*)d_in[1];   // [n*200, 512]
    const float* target = (const float*)d_in[2];   // [n*200]
    float* out = (float*)d_out;                    // scalar
    float* ws  = (float*)d_ws;                     // n floats of scratch

    const int n = in_sizes[0] / DDIM;              // 2048

    ssloss_persample<<<n, 128, 0, stream>>>(query, key, target, ws);
    ssloss_reduce_mean<<<1, 256, 0, stream>>>(ws, out, n);
}
